// RNNWithCheckpoint_26688926777979
// MI455X (gfx1250) — matrix operand order repaired and checked by reading, not compiled, not run on hardware
//
#include <hip/hip_runtime.h>

#define T_SEQ  512
#define BATCH  128
#define HID    1024
#define LAYERS 4

typedef __attribute__((ext_vector_type(16))) __bf16 v16bf;
typedef __attribute__((ext_vector_type(8)))  float  v8f;

union FragBF {
    v16bf v;
    unsigned short s16[16];
    uint4 q[2];
};

__device__ __forceinline__ unsigned short f2bf(float f) {
    unsigned int u = __builtin_bit_cast(unsigned int, f);
    unsigned int r = u + 0x7FFFu + ((u >> 16) & 1u);   // round-to-nearest-even
    return (unsigned short)(r >> 16);
}

__device__ __forceinline__ float fast_tanh(float x) {
    // tanh(x) = (e^{2x}-1)/(e^{2x}+1); e^{2x} = 2^{2*log2(e)*x}
    x = fminf(fmaxf(x, -8.0f), 8.0f);
    float e = __builtin_amdgcn_exp2f(x * 2.8853900817779268f);
    return (e - 1.0f) * __builtin_amdgcn_rcpf(e + 1.0f);
}

// ---------------------------------------------------------------------------
// One RNN cell step for one layer:
//   out = tanh(Xin @ Wih^T + Hprev @ Whh^T + bias)
// All operands bf16 (converted in prep). Split-K 4-way: the 2048-deep K space
// [Xin*Wih | Hprev*Whh] is cut into four 512-chunks, one wave per chunk per
// 16x16 output tile, partials combined through LDS.
// Workgroup: 8 waves = 2 output tiles x 4 K-slices.
// Grid: 8 m-tiles x 32 n-pairs = 256 workgroups = 2048 waves per step.
// ---------------------------------------------------------------------------
__global__ __launch_bounds__(256)
void rnn_step_gemm(const unsigned short* __restrict__ Xin,     // [BATCH,HID] bf16 layer input
                   const unsigned short* __restrict__ Hprev,   // [BATCH,HID] bf16
                   const unsigned short* __restrict__ Wih,     // [HID,HID]  bf16, row = out feature
                   const unsigned short* __restrict__ Whh,     // [HID,HID]  bf16
                   const float* __restrict__ bias,             // [HID] (b_ih + b_hh)
                   unsigned short* __restrict__ Hout,          // [BATCH,HID] bf16
                   float* __restrict__ Fout)                   // [BATCH,HID] f32 or null
{
    __shared__ float red[2][3][256];

    const int tid     = threadIdx.x;
    const int lane    = tid & 31;
    const int wave    = tid >> 5;     // 0..7
    const int tileIdx = wave >> 2;    // 0..1 : which 16x16 tile of the 16x32 strip
    const int slice   = wave & 3;     // 0..3 : K slice

    const int mTile  = blockIdx.x & 7;      // 8 tiles of M=128
    const int nGroup = blockIdx.x >> 3;     // 32 pairs of N=1024
    const int m0 = mTile * 16;
    const int n0 = nGroup * 32 + tileIdx * 16;

    // ISA 7.12.2 lane addressing for 16-bit A (16x32) and B (32x16) fragments
    const int aRow  = m0 + (lane & 15);
    const int aKoff = (lane >> 4) * 8;    // lanes 0-15: K 0-7(+16..23); 16-31: K 8-15(+24..31)
    const int bCol  = n0 + (lane & 15);
    const int bKoff = (lane >> 4) * 8;    // lanes 0-15: K 0-7(+16..23); 16-31: K 8-15(+24..31)

    // K-slice -> (A source, B source, k range). Uniform per wave, hoisted
    // out of the loop: no EXEC churn inside the hot loop.
    const unsigned short* Arow;
    const unsigned short* Brow;
    int k0;
    if (slice < 2) {
        Arow = Xin   + (size_t)aRow * HID;
        Brow = Wih   + (size_t)bCol * HID;   // B[k,n] = W[n,k]
        k0 = slice * 512;
    } else {
        Arow = Hprev + (size_t)aRow * HID;
        Brow = Whh   + (size_t)bCol * HID;
        k0 = (slice - 2) * 512;
    }

    v8f acc;
    for (int i = 0; i < 8; ++i) acc[i] = 0.0f;

    #pragma unroll 4
    for (int k = k0; k < k0 + 512; k += 32) {
        FragBF a, b;
        a.q[0] = *(const uint4*)(Arow + k + aKoff);
        a.q[1] = *(const uint4*)(Arow + k + aKoff + 16);
        b.q[0] = *(const uint4*)(Brow + k + bKoff);
        b.q[1] = *(const uint4*)(Brow + k + bKoff + 16);
        acc = __builtin_amdgcn_wmma_f32_16x16x32_bf16(
                  false, a.v, false, b.v, (short)0, acc, false, false);
    }

    // Split-K reduction through LDS; slice 0 combines + runs the epilogue.
    if (slice != 0) {
        float* dst = &red[tileIdx][slice - 1][0];
        for (int r = 0; r < 8; ++r)
            dst[((lane >> 4) * 8 + r) * 16 + (lane & 15)] = acc[r];
    }
    __syncthreads();
    if (slice == 0) {
        const int n  = n0 + (lane & 15);
        const float bv = bias[n];
        for (int r = 0; r < 8; ++r) {
            const int idx = ((lane >> 4) * 8 + r) * 16 + (lane & 15);
            float v = acc[r] + red[tileIdx][0][idx] + red[tileIdx][1][idx]
                             + red[tileIdx][2][idx] + bv;
            v = fast_tanh(v);
            const int m = m0 + (lane >> 4) * 8 + r;     // C/D layout: VGPR r -> row
            Hout[(size_t)m * HID + n] = f2bf(v);
            if (Fout) Fout[(size_t)m * HID + n] = v;
        }
    }
}

// ---------------------------------------------------------------------------
// Prep kernels (re-run every launch; deterministic)
// ---------------------------------------------------------------------------
__global__ void cvt_f32_bf16(const float* __restrict__ src,
                             unsigned short* __restrict__ dst, int n) {
    int i = blockIdx.x * blockDim.x + threadIdx.x;
    int stride = gridDim.x * blockDim.x;
    for (; i < n; i += stride) dst[i] = f2bf(src[i]);
}

__global__ void bias_sum(const float* __restrict__ a, const float* __restrict__ b,
                         float* __restrict__ o, int n) {
    int i = blockIdx.x * blockDim.x + threadIdx.x;
    if (i < n) o[i] = a[i] + b[i];
}

__global__ void zero_u16(unsigned short* __restrict__ p, int n) {
    int i = blockIdx.x * blockDim.x + threadIdx.x;
    int stride = gridDim.x * blockDim.x;
    for (; i < n; i += stride) p[i] = 0;
}

// ---------------------------------------------------------------------------
extern "C" void kernel_launch(void* const* d_in, const int* in_sizes, int n_in,
                              void* d_out, int out_size, void* d_ws, size_t ws_size,
                              hipStream_t stream) {
    (void)in_sizes; (void)n_in; (void)out_size;

    const float* x    = (const float*)d_in[0];   // [T, B, H]
    const float* W_ih = (const float*)d_in[1];   // [L, H, H]
    const float* W_hh = (const float*)d_in[2];   // [L, H, H]
    const float* b_ih = (const float*)d_in[3];   // [L, H]
    const float* b_hh = (const float*)d_in[4];   // [L, H]
    float* out = (float*)d_out;                  // [T, B, H]

    const size_t WSZ = (size_t)LAYERS * HID * HID;    // 4M elems per weight set
    const size_t HSZ = (size_t)LAYERS * BATCH * HID;  // 512K elems per h snapshot
    const size_t XT  = (size_t)BATCH * HID;           // elems per timestep

    unsigned short* wih_bf = (unsigned short*)d_ws;      //  8 MB
    unsigned short* whh_bf = wih_bf + WSZ;               //  8 MB
    unsigned short* hbuf   = whh_bf + WSZ;               //  2 MB (ping-pong)
    float*          bias   = (float*)(hbuf + 2 * HSZ);   // 16 KB
    unsigned short* xbf    = (unsigned short*)(bias + LAYERS * HID);

    const size_t base_bytes = (size_t)((char*)xbf - (char*)d_ws);
    const bool   full_x = ws_size >= base_bytes + (size_t)T_SEQ * XT * sizeof(unsigned short);

    cvt_f32_bf16<<<512, 256, 0, stream>>>(W_ih, wih_bf, (int)WSZ);
    cvt_f32_bf16<<<512, 256, 0, stream>>>(W_hh, whh_bf, (int)WSZ);
    bias_sum<<<(LAYERS * HID + 255) / 256, 256, 0, stream>>>(b_ih, b_hh, bias, LAYERS * HID);
    zero_u16<<<256, 256, 0, stream>>>(hbuf, (int)(2 * HSZ));
    if (full_x)   // one bulk conversion of all of x (stays off the sequential chain)
        cvt_f32_bf16<<<4096, 256, 0, stream>>>(x, xbf, (int)(T_SEQ * XT));

    for (int t = 0; t < T_SEQ; ++t) {
        const int cur = t & 1, nxt = cur ^ 1;

        const unsigned short* x_t;
        if (full_x) {
            x_t = xbf + (size_t)t * XT;
        } else {
            // small fallback buffer, converted just-in-time each step
            cvt_f32_bf16<<<128, 256, 0, stream>>>(x + (size_t)t * XT, xbf, (int)XT);
            x_t = xbf;
        }

        for (int l = 0; l < LAYERS; ++l) {
            const unsigned short* Ain = (l == 0)
                ? x_t
                : hbuf + (size_t)nxt * HSZ + (size_t)(l - 1) * XT;
            const unsigned short* Hp = hbuf + (size_t)cur * HSZ + (size_t)l * XT;
            unsigned short*       Ho = hbuf + (size_t)nxt * HSZ + (size_t)l * XT;
            const unsigned short* Wi = wih_bf + (size_t)l * HID * HID;
            const unsigned short* Wh = whh_bf + (size_t)l * HID * HID;
            const float*          bs = bias + l * HID;
            float* Fo = (l == LAYERS - 1) ? out + (size_t)t * XT : nullptr;

            rnn_step_gemm<<<256, 256, 0, stream>>>(Ain, Hp, Wi, Wh, bs, Ho, Fo);
        }
    }
}
